// BlockRAblation_86045374808449
// MI455X (gfx1250) — matrix operand order repaired and checked by reading, not compiled, not run on hardware
//
#include <hip/hip_runtime.h>

// Problem constants (match reference)
constexpr int Bb = 2, T = 512, E = 1024, H = 256;
constexpr int BT = Bb * T;
#define EPSF 1e-6f

typedef __attribute__((ext_vector_type(16))) _Float16 v16h;
typedef __attribute__((ext_vector_type(8)))  _Float16 v8h;
typedef __attribute__((ext_vector_type(8)))  float    v8f;
typedef unsigned int u32x4 __attribute__((ext_vector_type(4)));
typedef int          i32x8 __attribute__((ext_vector_type(8)));
typedef int          i32x4 __attribute__((ext_vector_type(4)));

// ---- weight swizzle: row-major fp32 [Kdim x Ndim] -> fragment-major f16 ----
// Fragment layout: frag t = (ntile*(Kdim/32) + ktile); lane l owns 16 halves,
// element e of lane l = w[ktile*32 + (l>>4)*16 + e][ntile*16 + (l&15)]
// => B fragment load in GEMM is one contiguous 32B read per lane.
__global__ void k_swizzle_w(const float* __restrict__ w, _Float16* __restrict__ o,
                            int Kdim, int Ndim) {
  const int idx = blockIdx.x * 256 + threadIdx.x;  // 0 .. Kdim*Ndim-1
  const int e = idx & 15;
  const int l = (idx >> 4) & 31;
  const int t = idx >> 9;
  const int ktiles = Kdim >> 5;
  const int n = t / ktiles, kt = t % ktiles;
  const int K = kt * 32 + ((l >> 4) * 16) + e;
  const int col = n * 16 + (l & 15);
  o[idx] = (_Float16)w[(size_t)K * Ndim + col];
}

// ---------------- RMS-norm of x over E, emit f16 r -------------------------
__global__ void k_norm_x(const float* __restrict__ x, _Float16* __restrict__ rh) {
  __shared__ float red[256];
  const int row = blockIdx.x;
  const int t = threadIdx.x;
  const float4 v = ((const float4*)(x + (size_t)row * E))[t];  // 256*4 = E
  red[t] = v.x * v.x + v.y * v.y + v.z * v.z + v.w * v.w;
  __syncthreads();
  for (int s = 128; s > 0; s >>= 1) {
    if (t < s) red[t] += red[t + s];
    __syncthreads();
  }
  const float scale = rsqrtf(red[0] / (float)E + EPSF);
  _Float16* rr = rh + (size_t)row * E;
  rr[4 * t + 0] = (_Float16)(v.x * scale);
  rr[4 * t + 1] = (_Float16)(v.y * scale);
  rr[4 * t + 2] = (_Float16)(v.z * scale);
  rr[4 * t + 3] = (_Float16)(v.w * scale);
}

// A-fragment (16-bit 16x32): lane l, element e -> K = 16*(l>>4) + e
__device__ __forceinline__ v16h load_afrag(const _Float16* arow, int g) {
  const v8h lo = *(const v8h*)(arow + g * 16);
  const v8h hi = *(const v8h*)(arow + g * 16 + 8);
  v16h af;
#pragma unroll
  for (int e = 0; e < 8; ++e) { af[e] = lo[e]; af[e + 8] = hi[e]; }
  return af;
}

// -------- fused dual GEMM: a = r@w1+b1, b = r@w2+b2 (WMMA f16, swizzled B) --
// grid = (H/16, BT/16), block = 32 (one wave per 16x16 tile, EXEC all ones)
__global__ void k_gemm_ab(const _Float16* __restrict__ rh,
                          const _Float16* __restrict__ w1s,
                          const _Float16* __restrict__ w2s,
                          const float* __restrict__ b1,
                          const float* __restrict__ b2,
                          float* __restrict__ aout,
                          float* __restrict__ bout) {
  const int lane = threadIdx.x & 31;
  const int g = lane >> 4, m = lane & 15;
  const int m0 = blockIdx.y * 16;
  const int col = blockIdx.x * 16 + m;
  const int ktiles = E / 32;
  const _Float16* arow = rh + (size_t)(m0 + m) * E;
  const v16h* w1t = (const v16h*)w1s + (size_t)blockIdx.x * ktiles * 32 + lane;
  const v16h* w2t = (const v16h*)w2s + (size_t)blockIdx.x * ktiles * 32 + lane;
  v8f acc1 = {}, acc2 = {};
  for (int kt = 0; kt < ktiles; ++kt) {
    const v16h af  = load_afrag(arow + kt * 32, g);
    const v16h bf1 = w1t[kt * 32];
    const v16h bf2 = w2t[kt * 32];
    acc1 = __builtin_amdgcn_wmma_f32_16x16x32_f16(false, af, false, bf1,
                                                  (short)0, acc1, false, false);
    acc2 = __builtin_amdgcn_wmma_f32_16x16x32_f16(false, af, false, bf2,
                                                  (short)0, acc2, false, false);
  }
  const float bias1 = b1[col], bias2 = b2[col];
#pragma unroll
  for (int r = 0; r < 8; ++r) {
    const int row = m0 + r + 8 * g;  // C/D layout: VGPR r, lane group g
    aout[(size_t)row * H + col] = acc1[r] + bias1;
    bout[(size_t)row * H + col] = acc2[r] + bias2;
  }
}

// ---------------- RMS-norm a and b over H (in place; rewritten each launch) --
__global__ void k_rownorm_H(float* __restrict__ a, float* __restrict__ b) {
  __shared__ float red[256];
  float* p = (blockIdx.y == 0 ? a : b) + (size_t)blockIdx.x * H;
  const int t = threadIdx.x;
  const float v = p[t];
  red[t] = v * v;
  __syncthreads();
  for (int s = 128; s > 0; s >>= 1) {
    if (t < s) red[t] += red[t + s];
    __syncthreads();
  }
  p[t] = v * rsqrtf(red[0] / (float)H + EPSF);
}

// ---------------- causal pairwise relu mean + fused post-norm ---------------
// block = output row i (256 threads = H features); j tiled via TDM into LDS.
#define TJ 32
__global__ void k_quad(const float* __restrict__ a, const float* __restrict__ b,
                       _Float16* __restrict__ r2h) {
  __shared__ float bj[TJ * H];   // 32 KB tile, loaded by Tensor Data Mover
  __shared__ float red[256];
  const int gi = blockIdx.x;     // 0..BT-1
  const int bb = gi / T, i = gi % T;
  const int t = threadIdx.x;     // feature h
  const float av = a[(size_t)gi * H + t];
  const float* brow = b + (size_t)bb * T * H;
  float acc = 0.f;
  for (int j0 = 0; j0 <= i; j0 += TJ) {
    const int nj = (i + 1 - j0 < TJ) ? (i + 1 - j0) : TJ;
    __syncthreads();             // previous tile fully consumed
    if (threadIdx.x < 32) {
      // ---- Tensor DMA: contiguous nj*H fp32 elements -> LDS (1-D tile) ----
      const unsigned n_el = (unsigned)nj * H;                 // <= 8192
      const unsigned lds  = (unsigned)(unsigned long long)(size_t)bj;
      const unsigned long long ga =
          (unsigned long long)(size_t)(brow + (size_t)j0 * H);
      u32x4 g0 = { 1u,                                        // count=1, user D#
                   lds,                                       // lds_addr[63:32]
                   (unsigned)ga,                              // global_addr lo
                   (unsigned)((ga >> 32) & 0x01FFFFFFu) | (2u << 30) }; // type=2
      i32x8 g1 = { (int)(2u << 16),                           // data_size=4B, mask=0
                   (int)((n_el & 0xFFFFu) << 16),             // tensor_dim0[15:0]
                   (int)((n_el >> 16) | (1u << 16)),          // td0[31:16], td1=1
                   (int)((n_el & 0xFFFFu) << 16),             // tile_dim0
                   0,                                         // tile_dim1/2 = 0
                   (int)n_el,                                 // dim0 stride lo
                   0, 0 };
      const i32x4 z4 = { 0, 0, 0, 0 };
      const i32x8 z8 = { 0, 0, 0, 0, 0, 0, 0, 0 };
      __builtin_amdgcn_tensor_load_to_lds(g0, g1, z4, z4, z8, 0);  // 6-arg form
      __builtin_amdgcn_s_wait_tensorcnt(0);                   // wave-private cnt
    }
    __syncthreads();             // tile visible to all 8 waves
#pragma unroll 4
    for (int jj = 0; jj < nj; ++jj)
      acc += fmaxf(av + bj[jj * H + t], 0.f);
  }
  const float y = acc / (float)(i + 1);
  red[t] = y * y;
  __syncthreads();
  for (int s = 128; s > 0; s >>= 1) {
    if (t < s) red[t] += red[t + s];
    __syncthreads();
  }
  r2h[(size_t)gi * H + t] = (_Float16)(y * rsqrtf(red[0] / (float)H + EPSF));
}

// ---------------- final GEMM: out = x + y@w3 + b3 (WMMA f16, swizzled B) ----
// grid = (E/16, BT/16), block = 32
__global__ void k_gemm_out(const _Float16* __restrict__ r2h,
                           const _Float16* __restrict__ w3s,
                           const float* __restrict__ b3,
                           const float* __restrict__ x,
                           float* __restrict__ out) {
  const int lane = threadIdx.x & 31;
  const int g = lane >> 4, m = lane & 15;
  const int m0 = blockIdx.y * 16;
  const int col = blockIdx.x * 16 + m;
  const int ktiles = H / 32;
  const _Float16* arow = r2h + (size_t)(m0 + m) * H;
  const v16h* w3t = (const v16h*)w3s + (size_t)blockIdx.x * ktiles * 32 + lane;
  v8f acc = {};
#pragma unroll
  for (int kt = 0; kt < ktiles; ++kt) {
    const v16h af = load_afrag(arow + kt * 32, g);
    const v16h bf = w3t[kt * 32];
    acc = __builtin_amdgcn_wmma_f32_16x16x32_f16(false, af, false, bf,
                                                 (short)0, acc, false, false);
  }
  const float bias = b3[col];
#pragma unroll
  for (int r = 0; r < 8; ++r) {
    const size_t idx = (size_t)(m0 + r + 8 * g) * E + col;
    out[idx] = x[idx] + acc[r] + bias;
  }
}

extern "C" void kernel_launch(void* const* d_in, const int* in_sizes, int n_in,
                              void* d_out, int out_size, void* d_ws, size_t ws_size,
                              hipStream_t stream) {
  (void)in_sizes; (void)n_in; (void)out_size; (void)ws_size;
  const float* x  = (const float*)d_in[0];
  const float* w1 = (const float*)d_in[1];
  const float* b1 = (const float*)d_in[2];
  const float* w2 = (const float*)d_in[3];
  const float* b2 = (const float*)d_in[4];
  const float* w3 = (const float*)d_in[5];
  const float* b3 = (const float*)d_in[6];
  float* out = (float*)d_out;

  // Workspace layout (bytes): total 6 MB
  char* ws = (char*)d_ws;
  _Float16* w1s = (_Float16*)(ws + (size_t)0);            // 512 KB (swizzled)
  _Float16* w2s = (_Float16*)(ws + ((size_t)512 << 10));  // 512 KB (swizzled)
  _Float16* w3s = (_Float16*)(ws + ((size_t)1024 << 10)); // 512 KB (swizzled)
  _Float16* rh  = (_Float16*)(ws + ((size_t)1536 << 10)); // 2 MB  (BT x E f16)
  float*    aa  = (float*)   (ws + ((size_t)3584 << 10)); // 1 MB  (BT x H f32)
  float*    bbf = (float*)   (ws + ((size_t)4608 << 10)); // 1 MB  (BT x H f32)
  _Float16* r2h = (_Float16*)(ws + ((size_t)5632 << 10)); // 512 KB (BT x H f16)

  k_swizzle_w<<<(E * H) / 256, 256, 0, stream>>>(w1, w1s, E, H);
  k_swizzle_w<<<(E * H) / 256, 256, 0, stream>>>(w2, w2s, E, H);
  k_swizzle_w<<<(H * E) / 256, 256, 0, stream>>>(w3, w3s, H, E);
  k_norm_x<<<BT, 256, 0, stream>>>(x, rh);
  k_gemm_ab<<<dim3(H / 16, BT / 16), 32, 0, stream>>>(rh, w1s, w2s, b1, b2, aa, bbf);
  k_rownorm_H<<<dim3(BT, 2), H, 0, stream>>>(aa, bbf);
  k_quad<<<BT, H, 0, stream>>>(aa, bbf, r2h);
  k_gemm_out<<<dim3(E / 16, BT / 16), 32, 0, stream>>>(r2h, w3s, b3, x, out);
}
